// VQVAE_7842610283123
// MI455X (gfx1250) — matrix operand order repaired and checked by reading, not compiled, not run on hardware
//
#include <hip/hip_runtime.h>

// ---------------- WMMA types / helpers ----------------
typedef __attribute__((ext_vector_type(16))) __bf16 v16bf;
typedef __attribute__((ext_vector_type(8)))  float  v8f;

union Frag { v16bf v; unsigned int u[8]; };

__device__ __forceinline__ unsigned short bf16bits(float f) {
    unsigned int u = __builtin_bit_cast(unsigned int, f);
    u += 0x7fffu + ((u >> 16) & 1u);            // round-to-nearest-even
    return (unsigned short)(u >> 16);
}
__device__ __forceinline__ float bf2f(unsigned short s) {
    return __builtin_bit_cast(float, (unsigned int)s << 16);
}
__device__ __forceinline__ unsigned int pk2bf(float a, float b) {
    return (unsigned int)bf16bits(a) | ((unsigned int)bf16bits(b) << 16);
}
// A-fragment: VGPR j holds 2 consecutive K; K0 per ISA 16-bit A 16x32 table
__device__ __forceinline__ int a_k0(int lane, int j) {
    return (j < 4 ? 2 * j : 2 * j + 8) + ((lane >= 16) ? 8 : 0);
}

#define H64 64
#define TLEN 5000
#define PNUM 156

// ---------------- Fused dilated-causal-conv (as 3-tap GEMM) + bias + LN + ReLU + residual ----
// Activations live in global memory as bf16. Input tile is staged to LDS with
// gfx1250 async global->LDS copies (ASYNCcnt); pad rows zero-filled via DS stores.
__global__ __launch_bounds__(256) void conv_ln_kernel(
    const unsigned short* __restrict__ in, unsigned short* __restrict__ out,
    const float* __restrict__ w,      // [taps][64][64] (k, cin, cout) fp32
    const float* __restrict__ bias,   // [64]
    const float* __restrict__ gamma, const float* __restrict__ beta,
    const unsigned short* __restrict__ res, const float* __restrict__ alpha_arr, int aidx,
    int taps, int dil, int do_ln, int do_relu, int do_res, int T)
{
    const int b   = blockIdx.y;
    const int t0  = blockIdx.x * 128;
    const int tid = threadIdx.x;
    const int wave = tid >> 5, lane = tid & 31;

    __shared__ unsigned short xt[144 * 64];     // input tile (bf16), 128 rows + up to 16 halo
    __shared__ float stage[8 * 16 * 64];        // 32KB; aliased as weight fragments during MMA
    __shared__ float sb[64], sg[64], sbe[64];
    __shared__ float smean[128], srs[128];
    unsigned short* wlds = (unsigned short*)stage;

    const int halo = (taps == 3) ? 2 * dil : 0;
    const int rows = 128 + halo;

    if (tid < 64) {
        sb[tid]  = bias[tid];
        sg[tid]  = do_ln ? gamma[tid] : 1.0f;
        sbe[tid] = do_ln ? beta[tid]  : 0.0f;
    }

    // ---- stage input tile: async global->LDS 16B chunks; zero-fill pad rows ----
    {
        const unsigned short* inb = in + (size_t)b * T * H64;
        const int nchunk = rows * 8;            // 8 x 16B per 128B row
        for (int ci = tid; ci < nchunk; ci += 256) {
            int r = ci >> 3;
            int t = t0 - halo + r;
            unsigned lds_off = (unsigned)(uintptr_t)(void*)&xt[ci * 8];
            if (t >= 0 && t < T) {
                unsigned long long ga =
                    (unsigned long long)(uintptr_t)(inb + (size_t)t * 64 + (ci & 7) * 8);
                asm volatile("global_load_async_to_lds_b128 %0, %1, off"
                             :: "v"(lds_off), "v"(ga) : "memory");
            } else {
                uint4 z = {0u, 0u, 0u, 0u};
                *(uint4*)&xt[ci * 8] = z;       // disjoint from async-loaded rows
            }
        }
    }
    // ---- stage weights pre-swizzled into B-fragment layout: frag f=(tap*2+kh)*4+nt ----
    const int nfrag = taps * 8;
    for (int i = tid; i < nfrag * 512; i += 256) {
        int f  = i >> 9;
        int ln_ = (i >> 4) & 31;
        int h  = i & 15;
        int tap = f >> 3, kh = (f >> 2) & 1, nt = f & 3;
        int K = (h < 8 ? h : h + 8) + ((ln_ >= 16) ? 8 : 0) + kh * 32;
        int N = nt * 16 + (ln_ & 15);
        wlds[i] = bf16bits(w[tap * 4096 + K * 64 + N]);
    }
    asm volatile("s_wait_asynccnt 0x0" ::: "memory");
    __syncthreads();

    v8f acc[4];
    #pragma unroll
    for (int nt = 0; nt < 4; ++nt)
        #pragma unroll
        for (int e = 0; e < 8; ++e) acc[nt][e] = 0.0f;

    const int mrow = wave * 16 + (lane & 15);
    for (int tap = 0; tap < taps; ++tap) {
        const int arow = mrow + ((taps == 3) ? tap * dil : 0);  // LDS row = local_t + tap*dil
        #pragma unroll
        for (int kh = 0; kh < 2; ++kh) {
            Frag A;
            #pragma unroll
            for (int j = 0; j < 8; ++j)
                A.u[j] = *(const unsigned int*)&xt[arow * 64 + kh * 32 + a_k0(lane, j)];
            #pragma unroll
            for (int nt = 0; nt < 4; ++nt) {
                const int f = (tap * 2 + kh) * 4 + nt;
                Frag B;
                const unsigned int* wp = (const unsigned int*)&wlds[(f * 32 + lane) * 16];
                #pragma unroll
                for (int j = 0; j < 8; ++j) B.u[j] = wp[j];
                acc[nt] = __builtin_amdgcn_wmma_f32_16x16x32_bf16(
                    false, A.v, false, B.v, (short)0, acc[nt], false, false);
            }
        }
    }

    __syncthreads();   // all waves done reading wlds -> safe to reuse as stage
    {
        float* st = stage + wave * 1024;
        #pragma unroll
        for (int nt = 0; nt < 4; ++nt)
            #pragma unroll
            for (int v = 0; v < 8; ++v) {
                int r = v + ((lane >= 16) ? 8 : 0);
                int c = nt * 16 + (lane & 15);
                st[r * 64 + c] = acc[nt][v] + sb[c];   // bias folded in before LN stats
            }
    }
    __syncthreads();

    // ---- LN stats: one thread per row ----
    if (tid < 128) {
        const float* rp = stage + (tid >> 4) * 1024 + (tid & 15) * 64;
        float m = 0.0f, s2 = 0.0f;
        for (int c = 0; c < 64; ++c) { float v = rp[c]; m += v; s2 += v * v; }
        m *= (1.0f / 64.0f);
        float var = s2 * (1.0f / 64.0f) - m * m;
        smean[tid] = m;
        srs[tid]   = rsqrtf(var + 1e-8f);
    }
    __syncthreads();

    // ---- coalesced epilogue store (bf16) ----
    const float alpha = do_res ? alpha_arr[aidx] : 0.0f;
    for (int i = tid; i < 128 * 64; i += 256) {
        int r = i >> 6, c = i & 63;
        int t = t0 + r;
        if (t < T) {
            float v = stage[(r >> 4) * 1024 + (r & 15) * 64 + c];
            if (do_ln)   v = (v - smean[r]) * srs[r] * sg[c] + sbe[c];
            if (do_relu) v = fmaxf(v, 0.0f);
            size_t gi = ((size_t)b * T + t) * 64 + c;
            if (do_res)  v = alpha * v + bf2f(res[gi]);
            out[gi] = bf16bits(v);
        }
    }
}

// ---------------- Encoder input projection: [B*T,12] @ [12,64] + b -> bf16 ----------------
__global__ __launch_bounds__(256) void encin_kernel(
    const float* __restrict__ x, const float* __restrict__ w,
    const float* __restrict__ bias, unsigned short* __restrict__ out)
{
    size_t i = (size_t)blockIdx.x * 256 + threadIdx.x;  // 160000*64
    size_t row = i >> 6; int c = (int)(i & 63);
    const float* xr = x + row * 12;
    float s = bias[c];
    #pragma unroll
    for (int k = 0; k < 12; ++k) s += xr[k] * w[k * 64 + c];
    out[i] = bf16bits(s);
}

// ---------------- quantize_input: WMMA GEMM [4992,2048] x qin_w^T -> quant fp32 ----------------
__global__ __launch_bounds__(256) void qin_kernel(
    const unsigned short* __restrict__ enc, const float* __restrict__ qw /*[64][2048]*/,
    const float* __restrict__ qb, float* __restrict__ quant)
{
    __shared__ unsigned short bl[4 * 32 * 16];
    const int tid = threadIdx.x, wave = tid >> 5, lane = tid & 31;
    const int rowbase = blockIdx.x * 128 + wave * 16;
    const int arow = rowbase + (lane & 15);
    const int ab = arow / PNUM, ap = arow % PNUM;
    const unsigned short* abase = enc + ((size_t)ab * TLEN + (size_t)ap * 32) * 64;

    v8f acc[4];
    #pragma unroll
    for (int nt = 0; nt < 4; ++nt)
        #pragma unroll
        for (int e = 0; e < 8; ++e) acc[nt][e] = 0.0f;

    for (int ks = 0; ks < 64; ++ks) {
        const int kbase = ks * 32;
        __syncthreads();
        for (int i = tid; i < 2048; i += 256) {
            int nt = i >> 9, ln_ = (i >> 4) & 31, h = i & 15;
            int K = kbase + (h < 8 ? h : h + 8) + ((ln_ >= 16) ? 8 : 0);
            int N = nt * 16 + (ln_ & 15);
            bl[i] = bf16bits(qw[(size_t)N * 2048 + K]);
        }
        __syncthreads();
        Frag A;
        #pragma unroll
        for (int j = 0; j < 8; ++j)   // enc already bf16: raw packed dword loads
            A.u[j] = *(const unsigned int*)&abase[kbase + a_k0(lane, j)];
        #pragma unroll
        for (int nt = 0; nt < 4; ++nt) {
            Frag B;
            const unsigned int* wp = (const unsigned int*)&bl[(nt * 32 + lane) * 16];
            #pragma unroll
            for (int j = 0; j < 8; ++j) B.u[j] = wp[j];
            acc[nt] = __builtin_amdgcn_wmma_f32_16x16x32_bf16(
                false, A.v, false, B.v, (short)0, acc[nt], false, false);
        }
    }
    #pragma unroll
    for (int nt = 0; nt < 4; ++nt)
        #pragma unroll
        for (int v = 0; v < 8; ++v) {
            int r = rowbase + v + ((lane >= 16) ? 8 : 0);
            int c = nt * 16 + (lane & 15);
            quant[(size_t)r * 64 + c] = acc[nt][v] + qb[c];
        }
}

// ---------------- codebook squared norms ----------------
__global__ void e2_kernel(const float* __restrict__ embed, float* __restrict__ e2)
{
    int c = blockIdx.x * 256 + threadIdx.x;
    if (c < 1024) {
        float s = 0.0f;
        for (int j = 0; j < 64; ++j) { float v = embed[j * 1024 + c]; s += v * v; }
        e2[c] = s;
    }
}

// ---------------- VQ nearest code + q + diff, one wave per row ----------------
__global__ __launch_bounds__(256) void vq_argmin_kernel(
    const float* __restrict__ quant, const float* __restrict__ embed,
    const float* __restrict__ e2, float* __restrict__ q,
    float* __restrict__ diff_acc, float* __restrict__ idx_out)
{
    __shared__ float fsh[8][64];
    __shared__ float rmin[8][32];
    __shared__ int   ridx[8][32];
    __shared__ int   bsh[8];
    const int tid = threadIdx.x, wave = tid >> 5, lane = tid & 31;
    const int row = blockIdx.x * 8 + wave;   // 624*8 = 4992 exact

    fsh[wave][lane]      = quant[(size_t)row * 64 + lane];
    fsh[wave][lane + 32] = quant[(size_t)row * 64 + lane + 32];

    float bestd = 3.4e38f; int besti = 0;
    for (int c = lane; c < 1024; c += 32) {
        float dot = 0.0f;
        for (int j = 0; j < 64; ++j) dot += fsh[wave][j] * embed[j * 1024 + c];
        float d = e2[c] - 2.0f * dot;
        if (d < bestd) { bestd = d; besti = c; }
    }
    rmin[wave][lane] = bestd; ridx[wave][lane] = besti;
    if (lane == 0) {
        for (int l = 1; l < 32; ++l)
            if (rmin[wave][l] < bestd) { bestd = rmin[wave][l]; besti = ridx[wave][l]; }
        bsh[wave] = besti;
        idx_out[row] = (float)besti;
    }
    const int best = bsh[wave];
    float dsum = 0.0f;
    for (int j = lane; j < 64; j += 32) {
        float qv = embed[j * 1024 + best];
        q[(size_t)row * 64 + j] = qv;               // quant_st == q in forward
        float df = qv - fsh[wave][j];
        dsum += df * df;
    }
    atomicAdd(diff_acc, dsum);
}

// ---------------- quantize_output: up[b,t,:] = qout_w[t,:] @ q_b + qout_b[t] -> bf16 ----------
__global__ __launch_bounds__(256) void qout_kernel(
    const float* __restrict__ q, const float* __restrict__ qw /*[5000][156]*/,
    const float* __restrict__ qob, unsigned short* __restrict__ up)
{
    __shared__ unsigned short bl[4 * 32 * 16];
    const int b = blockIdx.y, t0 = blockIdx.x * 128;
    const int tid = threadIdx.x, wave = tid >> 5, lane = tid & 31;
    const int mrow = t0 + wave * 16 + (lane & 15);

    v8f acc[4];
    #pragma unroll
    for (int nt = 0; nt < 4; ++nt)
        #pragma unroll
        for (int e = 0; e < 8; ++e) acc[nt][e] = 0.0f;

    for (int ks = 0; ks < 5; ++ks) {           // K = 156, padded to 160
        const int kbase = ks * 32;
        __syncthreads();
        for (int i = tid; i < 2048; i += 256) {
            int nt = i >> 9, ln_ = (i >> 4) & 31, h = i & 15;
            int K = kbase + (h < 8 ? h : h + 8) + ((ln_ >= 16) ? 8 : 0);
            int N = nt * 16 + (ln_ & 15);
            bl[i] = (K < PNUM) ? bf16bits(q[((size_t)b * PNUM + K) * 64 + N]) : (unsigned short)0;
        }
        __syncthreads();
        Frag A;
        #pragma unroll
        for (int j = 0; j < 8; ++j) {
            int p = kbase + a_k0(lane, j);
            float f0 = (mrow < TLEN && p     < PNUM) ? qw[(size_t)mrow * PNUM + p]     : 0.0f;
            float f1 = (mrow < TLEN && p + 1 < PNUM) ? qw[(size_t)mrow * PNUM + p + 1] : 0.0f;
            A.u[j] = pk2bf(f0, f1);
        }
        #pragma unroll
        for (int nt = 0; nt < 4; ++nt) {
            Frag B;
            const unsigned int* wp = (const unsigned int*)&bl[(nt * 32 + lane) * 16];
            #pragma unroll
            for (int j = 0; j < 8; ++j) B.u[j] = wp[j];
            acc[nt] = __builtin_amdgcn_wmma_f32_16x16x32_bf16(
                false, A.v, false, B.v, (short)0, acc[nt], false, false);
        }
    }
    #pragma unroll
    for (int nt = 0; nt < 4; ++nt)
        #pragma unroll
        for (int v = 0; v < 8; ++v) {
            int t = t0 + wave * 16 + v + ((lane >= 16) ? 8 : 0);
            if (t < TLEN) {
                int c = nt * 16 + (lane & 15);
                up[((size_t)b * TLEN + t) * 64 + c] = bf16bits(acc[nt][v] + qob[t]);
            }
        }
}

// ---------------- Decoder output projection: bf16 [B*T,64] @ [64,12] + b -> fp32 ----------------
__global__ __launch_bounds__(256) void decproj_kernel(
    const unsigned short* __restrict__ in, const float* __restrict__ w,
    const float* __restrict__ bias, float* __restrict__ out)
{
    size_t i = (size_t)blockIdx.x * 256 + threadIdx.x;  // 1,920,000
    size_t row = i / 12; int f = (int)(i % 12);
    const unsigned short* xr = in + row * 64;
    float s = bias[f];
    for (int j = 0; j < 64; ++j) s += bf2f(xr[j]) * w[j * 12 + f];
    out[i] = s;
}

__global__ void diff_final_kernel(const float* __restrict__ acc, float* __restrict__ diff_out)
{
    diff_out[0] = acc[0] * 1.25f / (4992.0f * 64.0f);   // (1 + BETA) * mean
}

// ---------------- host launch ----------------
extern "C" void kernel_launch(void* const* d_in, const int* in_sizes, int n_in,
                              void* d_out, int out_size, void* d_ws, size_t ws_size,
                              hipStream_t stream) {
    const float* x        = (const float*)d_in[0];
    const float* enc_in_w = (const float*)d_in[1];
    const float* enc_in_b = (const float*)d_in[2];
    const float* enc_c1w  = (const float*)d_in[3];
    const float* enc_c1b  = (const float*)d_in[4];
    const float* enc_l1g  = (const float*)d_in[5];
    const float* enc_l1b  = (const float*)d_in[6];
    const float* enc_c2w  = (const float*)d_in[7];
    const float* enc_c2b  = (const float*)d_in[8];
    const float* enc_l2g  = (const float*)d_in[9];
    const float* enc_l2b  = (const float*)d_in[10];
    const float* enc_a    = (const float*)d_in[11];
    const float* enc_ow   = (const float*)d_in[12];
    const float* enc_ob   = (const float*)d_in[13];
    const float* qin_w    = (const float*)d_in[14];
    const float* qin_b    = (const float*)d_in[15];
    const float* embed    = (const float*)d_in[16];
    const float* qout_w   = (const float*)d_in[17];
    const float* qout_b   = (const float*)d_in[18];
    const float* dec_c1w  = (const float*)d_in[19];
    const float* dec_c1b  = (const float*)d_in[20];
    const float* dec_l1g  = (const float*)d_in[21];
    const float* dec_l1b  = (const float*)d_in[22];
    const float* dec_c2w  = (const float*)d_in[23];
    const float* dec_c2b  = (const float*)d_in[24];
    const float* dec_l2g  = (const float*)d_in[25];
    const float* dec_l2b  = (const float*)d_in[26];
    const float* dec_a    = (const float*)d_in[27];
    const float* dec_ow   = (const float*)d_in[28];
    const float* dec_ob   = (const float*)d_in[29];
    const float* dpw      = (const float*)d_in[30];
    const float* dpb      = (const float*)d_in[31];

    // workspace layout (activations stored as bf16)
    unsigned short* bufA = (unsigned short*)d_ws;      // [B,T,64] bf16 = 20.48MB
    unsigned short* bufB = bufA + 10240000;            // [B,T,64] bf16
    float* qbuf    = (float*)(bufB + 10240000);        // [B,P,64] fp32
    float* quantb  = qbuf + 319488;                    // [B,P,64] fp32
    float* e2buf   = quantb + 319488;                  // [1024]
    float* diffacc = e2buf + 1024;                     // [1]

    float* dec_out  = (float*)d_out;               // 1,920,000
    float* diff_out = dec_out + 1920000;           // 1
    float* idx_out  = dec_out + 1920001;           // 4992

    hipMemsetAsync(diffacc, 0, sizeof(float), stream);

    const dim3 cgrid(40, 32);

    // -------- Encoder --------
    encin_kernel<<<40000, 256, 0, stream>>>(x, enc_in_w, enc_in_b, bufA);
    for (int i = 0; i < 8; ++i) {
        int d = (i & 1) ? 4 : 1;
        conv_ln_kernel<<<cgrid, 256, 0, stream>>>(
            bufA, bufB, enc_c1w + (size_t)i * 12288, enc_c1b + i * 64,
            enc_l1g + i * 64, enc_l1b + i * 64, nullptr, nullptr, 0,
            3, d, 1, 1, 0, TLEN);
        conv_ln_kernel<<<cgrid, 256, 0, stream>>>(
            bufB, bufA, enc_c2w + (size_t)i * 12288, enc_c2b + i * 64,
            enc_l2g + i * 64, enc_l2b + i * 64, bufA, enc_a, i,
            3, 2 * d, 1, 1, 1, TLEN);
    }
    conv_ln_kernel<<<cgrid, 256, 0, stream>>>(    // TCN output linear
        bufA, bufB, enc_ow, enc_ob, nullptr, nullptr, nullptr, nullptr, 0,
        1, 1, 0, 0, 0, TLEN);

    // -------- Quantizer --------
    qin_kernel<<<39, 256, 0, stream>>>(bufB, qin_w, qin_b, quantb);
    e2_kernel<<<4, 256, 0, stream>>>(embed, e2buf);
    vq_argmin_kernel<<<624, 256, 0, stream>>>(quantb, embed, e2buf, qbuf, diffacc, idx_out);
    qout_kernel<<<cgrid, 256, 0, stream>>>(qbuf, qout_w, qout_b, bufA);

    // -------- Decoder --------
    for (int i = 0; i < 8; ++i) {
        int d = (i & 1) ? 4 : 1;
        conv_ln_kernel<<<cgrid, 256, 0, stream>>>(
            bufA, bufB, dec_c1w + (size_t)i * 12288, dec_c1b + i * 64,
            dec_l1g + i * 64, dec_l1b + i * 64, nullptr, nullptr, 0,
            3, d, 1, 1, 0, TLEN);
        conv_ln_kernel<<<cgrid, 256, 0, stream>>>(
            bufB, bufA, dec_c2w + (size_t)i * 12288, dec_c2b + i * 64,
            dec_l2g + i * 64, dec_l2b + i * 64, bufA, dec_a, i,
            3, 2 * d, 1, 1, 1, TLEN);
    }
    conv_ln_kernel<<<cgrid, 256, 0, stream>>>(
        bufA, bufB, dec_ow, dec_ob, nullptr, nullptr, nullptr, nullptr, 0,
        1, 1, 0, 0, 0, TLEN);
    decproj_kernel<<<7500, 256, 0, stream>>>(bufB, dpw, dpb, dec_out);

    diff_final_kernel<<<1, 1, 0, stream>>>(diffacc, diff_out);
}
